// LSTM_57707180589181
// MI455X (gfx1250) — compile-verified, matrix operand order checked by reading, not run on hardware
//
#include <hip/hip_runtime.h>
#include <hip/hip_bf16.h>
#include <math.h>

// ---------------------------------------------------------------------------
// LSTM forward for MI455X (gfx1250, wave32, WMMA)
// gates = [h_prev | x_t] (64x1280) @ Wcat[g]^T (1024x1280)  via f16 WMMA.
// A-panel staged in LDS (320KB/WGP) once per block per step; B streams from L2.
// ---------------------------------------------------------------------------

typedef __attribute__((ext_vector_type(16))) _Float16 v16h;
typedef __attribute__((ext_vector_type(8)))  _Float16 v8h;
typedef __attribute__((ext_vector_type(8)))  float    v8f;

#define SEQ_LEN   512
#define BATCH     64
#define IN_SZ     256
#define HIDDEN    1024
#define KCAT      (HIDDEN + IN_SZ)          // 1280
#define KPAD      (KCAT + 8)                // 1288: bank-conflict-free LDS stride
#define NGATES    4

// workspace layout (bytes)
#define WCAT_ELEMS  ((size_t)NGATES * HIDDEN * KCAT)         // 5,242,880 f16
#define XH_ELEMS    ((size_t)SEQ_LEN * BATCH * IN_SZ)        // 8,388,608 f16
#define BIAS_ELEMS  ((size_t)NGATES * HIDDEN)                // 4096 f32
#define H_ELEMS     ((size_t)BATCH * HIDDEN)                 // 65,536

#define WCAT_BYTES  (WCAT_ELEMS * 2)
#define XH_BYTES    (XH_ELEMS * 2)
#define BIAS_BYTES  (BIAS_ELEMS * 4)
#define H_BYTES     (H_ELEMS * 2)
#define C_BYTES     (H_ELEMS * 4)

// ---------------------------------------------------------------------------
// Kernel 1: convert weights/x to f16, fold biases, zero h/c state.
// ---------------------------------------------------------------------------
__global__ void lstm_convert_kernel(
    const float* __restrict__ x,
    const float* __restrict__ Wfx, const float* __restrict__ bfx,
    const float* __restrict__ Wix, const float* __restrict__ bix,
    const float* __restrict__ Wgx, const float* __restrict__ bgx,
    const float* __restrict__ Wox, const float* __restrict__ box_,
    const float* __restrict__ Wfh, const float* __restrict__ bfh,
    const float* __restrict__ Wih, const float* __restrict__ bih,
    const float* __restrict__ Wgh, const float* __restrict__ bgh,
    const float* __restrict__ Woh, const float* __restrict__ boh,
    _Float16* __restrict__ Wcat, _Float16* __restrict__ xh,
    float* __restrict__ bias,
    _Float16* __restrict__ h0, _Float16* __restrict__ h1,
    float* __restrict__ c)
{
    const size_t tid = (size_t)blockIdx.x * blockDim.x + threadIdx.x;
    const size_t nth = (size_t)gridDim.x * blockDim.x;

    // Wcat[g][n][k] : k<1024 -> Wh_g[n][k] ; else Wx_g[n][k-1024]
    for (size_t i = tid; i < WCAT_ELEMS; i += nth) {
        int g = (int)(i / ((size_t)HIDDEN * KCAT));
        size_t rem = i % ((size_t)HIDDEN * KCAT);
        int n = (int)(rem / KCAT);
        int k = (int)(rem % KCAT);
        const float* Wh = (g == 0) ? Wfh : (g == 1) ? Wih : (g == 2) ? Wgh : Woh;
        const float* Wx = (g == 0) ? Wfx : (g == 1) ? Wix : (g == 2) ? Wgx : Wox;
        float v = (k < HIDDEN) ? Wh[(size_t)n * HIDDEN + k]
                               : Wx[(size_t)n * IN_SZ + (k - HIDDEN)];
        Wcat[i] = (_Float16)v;
    }
    // x -> f16
    for (size_t i = tid; i < XH_ELEMS; i += nth)
        xh[i] = (_Float16)x[i];
    // bias[g][n] = bh + bx
    for (size_t i = tid; i < BIAS_ELEMS; i += nth) {
        int g = (int)(i >> 10);
        int n = (int)(i & (HIDDEN - 1));
        const float* bh = (g == 0) ? bfh : (g == 1) ? bih : (g == 2) ? bgh : boh;
        const float* bx = (g == 0) ? bfx : (g == 1) ? bix : (g == 2) ? bgx : box_;
        bias[i] = bh[n] + bx[n];
    }
    // zero state
    for (size_t i = tid; i < H_ELEMS; i += nth) {
        h0[i] = (_Float16)0.f;
        h1[i] = (_Float16)0.f;
        c[i]  = 0.f;
    }
}

// ---------------------------------------------------------------------------
// Kernel 2: one LSTM time-step.
// grid: 64 blocks (one per 16-column tile of HIDDEN)
// block: 512 threads = 16 wave32 = 4 gates x 4 M-tiles (M=64 batch rows)
// A-panel [h|x_t] (64x1280 f16) staged in LDS (padded stride 1288), shared by
// all 16 waves; each wave runs 40 x v_wmma_f32_16x16x32_f16 over K=1280.
// Gates combined through LDS, fused elementwise c/h update.
// ---------------------------------------------------------------------------
__global__ __launch_bounds__(512, 1) void lstm_step_kernel(
    const _Float16* __restrict__ Wcat,   // [4][1024][1280]
    const float*    __restrict__ bias,   // [4][1024]
    const _Float16* __restrict__ xh,     // [512][64][256]
    const _Float16* __restrict__ h_in,   // [64][1024] f16
    _Float16*       __restrict__ h_out,  // [64][1024] f16
    float*          __restrict__ c,      // [64][1024] f32 (in-place)
    float*          __restrict__ h_seq,  // d_out: [512][64][1024] f32
    int t)
{
    __shared__ _Float16 Atile[BATCH][KPAD];    // 164,864 B  (A-panel, padded)
    __shared__ float    pre[NGATES][BATCH][16]; // 16,384 B  (pre-activations)

    const int n0    = blockIdx.x * 16;             // column tile
    const int wave  = threadIdx.x >> 5;            // 0..15
    const int gate  = wave >> 2;                   // 0..3 (f,i,g,o)
    const int M0    = (wave & 3) * 16;             // batch-row tile
    const int lane  = threadIdx.x & 31;
    const int lrow  = lane & 15;
    const int koff  = (lane & 16) ? 8 : 0;

    const _Float16* x_t = xh + (size_t)t * BATCH * IN_SZ;

    // ---- stage [h_prev | x_t] into LDS (16B chunks, conflict-free) ----
    {
        const int nchunk = BATCH * (KCAT / 8);     // 10240 chunks of 8 f16
        for (int ck = threadIdx.x; ck < nchunk; ck += 512) {
            const int row = ck / (KCAT / 8);
            const int kc  = (ck % (KCAT / 8)) * 8;
            v8h v;
            if (kc < HIDDEN) v = *(const v8h*)(h_in + (size_t)row * HIDDEN + kc);
            else             v = *(const v8h*)(x_t + (size_t)row * IN_SZ + (kc - HIDDEN));
            *(v8h*)(&Atile[row][kc]) = v;
        }
    }
    __syncthreads();

    // ---- loop-invariant per-lane base pointers (immediate-offset loads) ----
    // WMMA 16-bit operand layout: lanes 0-15: row, K {k..k+7, k+16..k+23};
    // lanes 16-31: row, K {k+8..k+15, k+24..k+31}.
    const _Float16* pA = &Atile[M0 + lrow][koff];                        // LDS
    const _Float16* pB = Wcat + ((size_t)gate * HIDDEN + n0 + lrow) * KCAT + koff;

    v8f acc = {};
#pragma unroll 8
    for (int kk = 0; kk < KCAT; kk += 32) {
        v8h alo = *(const v8h*)(pA + kk);          // ds_load_b128
        v8h ahi = *(const v8h*)(pA + kk + 16);
        v8h blo = *(const v8h*)(pB + kk);          // global_load_b128
        v8h bhi = *(const v8h*)(pB + kk + 16);
        if (kk + 32 < KCAT)
            __builtin_prefetch(pB + kk + 32, 0, 1);  // global_prefetch_b8
        v16h a, b;
#pragma unroll
        for (int i = 0; i < 8; ++i) {
            a[i] = alo[i]; a[8 + i] = ahi[i];
            b[i] = blo[i]; b[8 + i] = bhi[i];
        }
        acc = __builtin_amdgcn_wmma_f32_16x16x32_f16(
                  false, a, false, b, (short)0, acc, false, false);
    }

    // ---- scatter pre-activations to LDS with bias added ----
    // C/D layout: VGPR r, lanes 0-15 -> M=r, N=lane; lanes 16-31 -> M=r+8.
    {
        const int mbase = M0 + ((lane & 16) ? 8 : 0);
        const float bv  = bias[(size_t)gate * HIDDEN + n0 + lrow];
#pragma unroll
        for (int r = 0; r < 8; ++r)
            pre[gate][mbase + r][lrow] = acc[r] + bv;
    }
    __syncthreads();

    // ---- fused elementwise LSTM update: 64x16 = 1024 elems, 2/thread ----
#pragma unroll
    for (int e = threadIdx.x; e < BATCH * 16; e += 512) {
        const int m = e >> 4;            // batch row
        const int n = e & 15;            // local column
        const int col = n0 + n;
        const float pf = pre[0][m][n];
        const float pi = pre[1][m][n];
        const float pg = pre[2][m][n];
        const float po = pre[3][m][n];
        const float fg = 1.f / (1.f + __expf(-pf));
        const float ig = 1.f / (1.f + __expf(-pi));
        const float og = 1.f / (1.f + __expf(-po));
        const float gg = tanhf(pg);
        const size_t sidx = (size_t)m * HIDDEN + col;
        const float cn = c[sidx] * fg + ig * gg;
        const float hn = tanhf(cn) * og;
        c[sidx] = cn;
        h_seq[(size_t)t * BATCH * HIDDEN + sidx] = hn;
        h_out[sidx] = (_Float16)hn;
    }
}

// ---------------------------------------------------------------------------
// Kernel 3: emit h_final (= h_seq[511]) and c_final after h_seq in d_out.
// ---------------------------------------------------------------------------
__global__ void lstm_finalize_kernel(const float* __restrict__ c,
                                     float* __restrict__ out)
{
    const int i = blockIdx.x * blockDim.x + threadIdx.x;
    if (i < (int)H_ELEMS) {
        const size_t HSEQ = (size_t)SEQ_LEN * BATCH * HIDDEN;
        out[HSEQ + i]           = out[(size_t)(SEQ_LEN - 1) * BATCH * HIDDEN + i];
        out[HSEQ + H_ELEMS + i] = c[i];
    }
}

// ---------------------------------------------------------------------------
extern "C" void kernel_launch(void* const* d_in, const int* in_sizes, int n_in,
                              void* d_out, int out_size, void* d_ws, size_t ws_size,
                              hipStream_t stream) {
    const float* x   = (const float*)d_in[0];
    const float* Wfx = (const float*)d_in[1];  const float* bfx = (const float*)d_in[2];
    const float* Wix = (const float*)d_in[3];  const float* bix = (const float*)d_in[4];
    const float* Wgx = (const float*)d_in[5];  const float* bgx = (const float*)d_in[6];
    const float* Wox = (const float*)d_in[7];  const float* box_ = (const float*)d_in[8];
    const float* Wfh = (const float*)d_in[9];  const float* bfh = (const float*)d_in[10];
    const float* Wih = (const float*)d_in[11]; const float* bih = (const float*)d_in[12];
    const float* Wgh = (const float*)d_in[13]; const float* bgh = (const float*)d_in[14];
    const float* Woh = (const float*)d_in[15]; const float* boh = (const float*)d_in[16];

    char* ws = (char*)d_ws;
    size_t off = 0;
    _Float16* Wcat = (_Float16*)(ws + off); off += WCAT_BYTES;
    _Float16* xh   = (_Float16*)(ws + off); off += XH_BYTES;
    float*    bias = (float*)   (ws + off); off += BIAS_BYTES;
    _Float16* h0   = (_Float16*)(ws + off); off += H_BYTES;
    _Float16* h1   = (_Float16*)(ws + off); off += H_BYTES;
    float*    c    = (float*)   (ws + off); off += C_BYTES;

    lstm_convert_kernel<<<4096, 256, 0, stream>>>(
        x, Wfx, bfx, Wix, bix, Wgx, bgx, Wox, box_,
        Wfh, bfh, Wih, bih, Wgh, bgh, Woh, boh,
        Wcat, xh, bias, h0, h1, c);

    float* out = (float*)d_out;
    for (int t = 0; t < SEQ_LEN; ++t) {
        const _Float16* hin = (t & 1) ? h1 : h0;
        _Float16*       hout = (t & 1) ? h0 : h1;
        lstm_step_kernel<<<HIDDEN / 16, 512, 0, stream>>>(
            Wcat, bias, xh, hin, hout, c, out, t);
    }

    lstm_finalize_kernel<<<(int)(H_ELEMS + 255) / 256, 256, 0, stream>>>(c, out);
}
